// EdgeConvBlock_72533407695116
// MI455X (gfx1250) — matrix operand order repaired and checked by reading, not compiled, not run on hardware
//
#include <hip/hip_runtime.h>

#define Bsz 8
#define Csz 64
#define Nsz 4096
#define Ksz 16
#define Gsz 64

typedef __attribute__((ext_vector_type(16))) __bf16 v16bf;
typedef __attribute__((ext_vector_type(8)))  __bf16 v8bf;
typedef __attribute__((ext_vector_type(8)))  float  v8f;

union FragA { v16bf v; v8bf h[2]; };

__device__ __forceinline__ __bf16 f2bf(float f) {
    unsigned int u = __builtin_bit_cast(unsigned int, f);
    unsigned int r = u + 0x7FFFu + ((u >> 16) & 1u);   // round-to-nearest-even
    unsigned short s = (unsigned short)(r >> 16);
    return __builtin_bit_cast(__bf16, s);
}

// ---------------------------------------------------------------------------
// Kernel 1: brute-force KNN (top-16 smallest d2, self included, sorted asc).
// One thread per query point; candidate coords tiled through LDS.
// ---------------------------------------------------------------------------
__global__ __launch_bounds__(256) void knn_kernel(
    const float* __restrict__ xyz,   // [B,3,N]
    int* __restrict__ knn)           // [B,N,16]
{
    __shared__ float sx[256], sy[256], sz[256];
    const int tid = threadIdx.x;
    const int b   = blockIdx.y;
    const int n   = blockIdx.x * 256 + tid;
    const float* base = xyz + (size_t)b * 3 * Nsz;

    const float qx = base[n];
    const float qy = base[Nsz + n];
    const float qz = base[2 * Nsz + n];

    float bd[16];
    int   bi[16];
#pragma unroll
    for (int j = 0; j < 16; ++j) { bd[j] = 3.4e38f; bi[j] = 0; }

    for (int t = 0; t < Nsz; t += 256) {
        __syncthreads();
        sx[tid] = base[t + tid];
        sy[tid] = base[Nsz + t + tid];
        sz[tid] = base[2 * Nsz + t + tid];
        __syncthreads();
        if (t + 256 < Nsz) {
            __builtin_prefetch(&base[t + 256 + tid], 0, 1);
            __builtin_prefetch(&base[Nsz + t + 256 + tid], 0, 1);
            __builtin_prefetch(&base[2 * Nsz + t + 256 + tid], 0, 1);
        }
        for (int m = 0; m < 256; ++m) {
            const float dx = qx - sx[m];
            const float dy = qy - sy[m];
            const float dz = qz - sz[m];
            const float d  = dx * dx + dy * dy + dz * dz;
            if (d < bd[15]) {
                bd[15] = d; bi[15] = t + m;
                // one unrolled bubble pass restores sorted order
#pragma unroll
                for (int j = 15; j > 0; --j) {
                    if (bd[j] < bd[j - 1]) {
                        float td = bd[j]; bd[j] = bd[j - 1]; bd[j - 1] = td;
                        int   ti = bi[j]; bi[j] = bi[j - 1]; bi[j - 1] = ti;
                    }
                }
            }
        }
    }

    int* o = knn + ((size_t)b * Nsz + n) * Ksz;
#pragma unroll
    for (int j = 0; j < 16; ++j) o[j] = bi[j];
}

// ---------------------------------------------------------------------------
// Kernel 2: fused gather + (W1-W2)*central + W2*knn GEMM (bf16 WMMA, f32 acc)
//           + folded bias/BN scale-shift + ReLU.
// 256 threads = 8 waves; wave w handles point n = n0+w (16 neighbor columns)
// and all four 16-row M tiles of G=64.
// ---------------------------------------------------------------------------
__global__ __launch_bounds__(256) void edge_gemm_kernel(
    const float* __restrict__ feat,    // [B,C,N]
    const float* __restrict__ weight,  // [G,2C]
    const float* __restrict__ bias,    // [G]
    const float* __restrict__ gamma,   // [G]
    const float* __restrict__ beta,    // [G]
    const float* __restrict__ rmean,   // [G]
    const float* __restrict__ rvar,    // [G]
    const int*   __restrict__ knn,     // [B,N,16]
    float* __restrict__ out)           // [B,G,N,16]
{
    __shared__ __align__(64) __bf16 sWA[Gsz][Csz];      // W1 - W2
    __shared__ __align__(64) __bf16 sWB[Gsz][Csz];      // W2
    __shared__ __align__(64) __bf16 sKnn[8][Ksz][Csz];  // gathered features
    __shared__ __align__(64) __bf16 sCen[8][Csz];       // central features
    __shared__ float sScale[Gsz];
    __shared__ float sShift[Gsz];
    __shared__ int   sIdx[8][Ksz];

    const int tid = threadIdx.x;
    const int b   = blockIdx.y;
    const int n0  = blockIdx.x * 8;

    // ---- stage transformed weights ----
    for (int i = tid; i < Gsz * Csz; i += 256) {
        const int g = i >> 6, c = i & 63;
        const float w1 = weight[g * (2 * Csz) + c];
        const float w2 = weight[g * (2 * Csz) + Csz + c];
        sWA[g][c] = f2bf(w1 - w2);
        sWB[g][c] = f2bf(w2);
    }
    // ---- fold bias + BN into scale/shift ----
    if (tid < Gsz) {
        const float inv = gamma[tid] * rsqrtf(rvar[tid] + 1e-5f);
        sScale[tid] = inv;
        sShift[tid] = beta[tid] + (bias[tid] - rmean[tid]) * inv;
    }
    // ---- neighbor indices for this block's 8 points ----
    if (tid < 128) {
        const int nl = tid >> 4, col = tid & 15;
        sIdx[nl][col] = knn[((size_t)b * Nsz + n0 + nl) * Ksz + col];
    }
    __syncthreads();

    // ---- gather neighbor feature columns into LDS (bf16) ----
    {
        const int p    = tid & 127;       // (nl,col) pair
        const int half = tid >> 7;        // channel half
        const int nl = p >> 4, col = p & 15;
        const int j  = sIdx[nl][col];
        const float* fp = feat + (size_t)b * Csz * Nsz + j;
        const int c0 = half * 32;
#pragma unroll 4
        for (int c = c0; c < c0 + 32; ++c)
            sKnn[nl][col][c] = f2bf(fp[(size_t)c * Nsz]);
    }
    // ---- central feature columns ----
    {
        const int nl = tid >> 5;
        const int c2 = (tid & 31) * 2;
        const float* fp = feat + (size_t)b * Csz * Nsz + (n0 + nl);
        sCen[nl][c2]     = f2bf(fp[(size_t)c2 * Nsz]);
        sCen[nl][c2 + 1] = f2bf(fp[(size_t)(c2 + 1) * Nsz]);
    }
    __syncthreads();

    // ---- WMMA main loop: D = (W1-W2)*central + W2*knn ----
    const int lane  = tid & 31;
    const int wv    = tid >> 5;            // local point index 0..7
    const int n     = n0 + wv;
    const int kbase = (lane >> 4) << 4;    // B-frag K half: 0 or 16
    const int kb    = (lane >> 4) << 3;    // unused; A-frag also uses kbase (0 or 16)
    const int mrow  = lane & 15;           // A-frag row within tile
    const int col   = lane & 15;           // B/D column (neighbor k)

    v8f acc[4] = {};

#pragma unroll
    for (int kc = 0; kc < Csz; kc += 32) {
        const v16bf bk = *(const v16bf*)&sKnn[wv][col][kc + kbase];
        const v16bf bc = *(const v16bf*)&sCen[wv][kc + kbase];
#pragma unroll
        for (int mt = 0; mt < 4; ++mt) {
            const int g = mt * 16 + mrow;
            FragA fa, fb;
            fa.h[0] = *(const v8bf*)&sWA[g][kc + kbase];
            fa.h[1] = *(const v8bf*)&sWA[g][kc + kbase + 8];
            fb.h[0] = *(const v8bf*)&sWB[g][kc + kbase];
            fb.h[1] = *(const v8bf*)&sWB[g][kc + kbase + 8];
            acc[mt] = __builtin_amdgcn_wmma_f32_16x16x32_bf16(
                false, fa.v, false, bc, (short)0, acc[mt], false, false);
            acc[mt] = __builtin_amdgcn_wmma_f32_16x16x32_bf16(
                false, fb.v, false, bk, (short)0, acc[mt], false, false);
        }
    }

    // ---- fused scale/shift + ReLU epilogue, k-contiguous stores ----
    const int mhi = (lane >> 4) * 8;
    float* op = out + (size_t)b * Gsz * Nsz * Ksz + (size_t)n * Ksz + col;
#pragma unroll
    for (int mt = 0; mt < 4; ++mt) {
#pragma unroll
        for (int r = 0; r < 8; ++r) {
            const int g = mt * 16 + r + mhi;
            float v = acc[mt][r] * sScale[g] + sShift[g];
            v = v > 0.f ? v : 0.f;
            op[(size_t)g * (Nsz * Ksz)] = v;
        }
    }
}

extern "C" void kernel_launch(void* const* d_in, const int* in_sizes, int n_in,
                              void* d_out, int out_size, void* d_ws, size_t ws_size,
                              hipStream_t stream) {
    const float* xyz    = (const float*)d_in[0];
    const float* feat   = (const float*)d_in[1];
    const float* weight = (const float*)d_in[2];
    const float* bias   = (const float*)d_in[3];
    const float* gamma  = (const float*)d_in[4];
    const float* beta   = (const float*)d_in[5];
    const float* rmean  = (const float*)d_in[6];
    const float* rvar   = (const float*)d_in[7];
    // d_in[8] = k (compile-time 16)

    int*   knn = (int*)d_ws;           // B*N*16 ints = 2 MB scratch
    float* out = (float*)d_out;

    knn_kernel<<<dim3(Nsz / 256, Bsz), 256, 0, stream>>>(xyz, knn);
    edge_gemm_kernel<<<dim3(Nsz / 8, Bsz), 256, 0, stream>>>(
        feat, weight, bias, gamma, beta, rmean, rvar, knn, out);
}
